// Attention_2173253452444
// MI455X (gfx1250) — matrix operand order repaired and checked by reading, not compiled, not run on hardware
//
#include <hip/hip_runtime.h>
#include <hip/hip_bf16.h>

// ---------- types ----------
typedef __bf16 bf16_t;
typedef __attribute__((ext_vector_type(16))) __bf16 bf16x16;
typedef __attribute__((ext_vector_type(4)))  __bf16 bf16x4;
typedef __attribute__((ext_vector_type(8)))  float  f32x8;
typedef __attribute__((ext_vector_type(4)))  unsigned int u32x4;
typedef __attribute__((ext_vector_type(8)))  int   i32x8;
typedef __attribute__((ext_vector_type(4)))  int   i32x4;

// ---------- problem dims ----------
#define B_DIM 32
#define NQ    1024
#define NK    1024
#define E_DIM 1024

// ---------- GEMM tiling ----------
#define BM 64
#define BN 256
#define BK 32
#define LDSS 40   // LDS row stride in bf16 (80B) = 32 data + 8 pad (TDM pad feature)

union FragU { bf16x16 f; uint4 u[2]; };

// Issue one 2D TDM tile load: global (row-major, row stride = stride_elems bf16)
// -> LDS at lds_off, tile_w x tile_h bf16 elements, padded 8 bf16 per 32-elem row
// so the LDS row stride is LDSS(=40) bf16.  D# layout per CDNA5 ISA 08 §8.3-8.5.
__device__ __forceinline__ void tdm_load_2d(unsigned lds_off, const bf16_t* gptr,
                                            unsigned tile_w, unsigned tile_h,
                                            unsigned long long stride_elems)
{
    const unsigned long long ga = (unsigned long long)(const void*)gptr;
    u32x4 g0;
    g0.x = 1u;                                   // count=1 (valid user descriptor)
    g0.y = lds_off;                              // lds_addr (bytes)
    g0.z = (unsigned)(ga & 0xffffffffu);         // global_addr[31:0]
    g0.w = (unsigned)((ga >> 32) & 0x01ffffffu)  // global_addr[56:32]
         | (2u << 30);                           // type = 2 ("image")

    i32x8 g1;
    g1[0] = (1 << 16)                            // data_size = 2 bytes
          | (1 << 20)                            // pad_enable
          | (3 << 22)                            // pad_interval code 3 = 16 DWORDs (64B)
          | (3 << 25);                           // pad_amount  code 3 = 4 DWORDs (8 bf16)
    g1[1] = (int)((tile_w & 0xffffu) << 16);     // tensor_dim0[15:0]  (tile-relative)
    g1[2] = (int)(((tile_w >> 16) & 0xffffu) | ((tile_h & 0xffffu) << 16)); // dim0 hi | dim1 lo
    g1[3] = (int)(((tile_h >> 16) & 0xffffu) | ((tile_w & 0xffffu) << 16)); // dim1 hi | tile_dim0
    g1[4] = (int)(tile_h & 0xffffu);             // tile_dim1 ; tile_dim2 = 0
    g1[5] = (int)(stride_elems & 0xffffffffu);   // tensor_dim0_stride[31:0]
    g1[6] = (int)((stride_elems >> 32) & 0xffffu); // dim0_stride[47:32]; dim1_stride[15:0]=0
    g1[7] = 0;

    const i32x4 gz4 = {0, 0, 0, 0};
    const i32x8 gz8 = {0, 0, 0, 0, 0, 0, 0, 0};
    // 6-arg form (clang-23 / therock-10.0 headers): groups 2/3 unused (2D tensor)
    __builtin_amdgcn_tensor_load_to_lds(g0, g1, gz4, gz4, gz8, 0);
}

__device__ __forceinline__ unsigned lds_off_of(const void* p)
{
    // flat shared-address low 32 bits == LDS byte offset (CDNA5 aperture mapping)
    return (unsigned)(unsigned long long)p;
}

// C[M,N] = A[M,K] * B[N,K]^T  (row-major, bf16 in, f32 accumulate)
// 256-thread block computes a 64x256 tile; each of 8 waves a 32x64 tile via
// 2x4 v_wmma_f32_16x16x32_bf16 accumulators.  A/B tiles staged by the Tensor
// Data Mover into double-buffered, pad-skewed LDS; TENSORcnt-synchronized.
template <typename OutT>
__global__ __launch_bounds__(256)
void gemm_nt_wmma(const bf16_t* __restrict__ A, const bf16_t* __restrict__ Bm,
                  OutT* __restrict__ C, int M, int N, int K,
                  long long sA, long long sB, long long sC)
{
    __shared__ bf16_t As[2][BM * LDSS];
    __shared__ bf16_t Bs[2][BN * LDSS];

    const int bz = blockIdx.z;
    A  += (long long)bz * sA;
    Bm += (long long)bz * sB;
    C  += (long long)bz * sC;

    const int m_wg = blockIdx.y * BM;
    const int n_wg = blockIdx.x * BN;

    const int tid    = threadIdx.x;
    const int lane   = tid & 31;
    const int wave   = tid >> 5;
    const int m_wave = (wave & 1) * 32;    // 0 / 32
    const int n_wave = (wave >> 1) * 64;   // 0 / 64 / 128 / 192

    const int fm = lane & 15;              // fragment row/col within 16x16 tile
    const int fh = lane >> 4;              // lane-half selector

    f32x8 acc[2][4];
    const f32x8 zero = {0.f, 0.f, 0.f, 0.f, 0.f, 0.f, 0.f, 0.f};
#pragma unroll
    for (int i = 0; i < 2; ++i)
#pragma unroll
        for (int j = 0; j < 4; ++j) acc[i][j] = zero;

    const bf16_t* a_tile = A  + (size_t)m_wg * K;
    const bf16_t* b_tile = Bm + (size_t)n_wg * K;

    const unsigned asOff[2] = { lds_off_of(&As[0][0]), lds_off_of(&As[1][0]) };
    const unsigned bsOff[2] = { lds_off_of(&Bs[0][0]), lds_off_of(&Bs[1][0]) };

    const int nsteps = K / BK;

    // prologue: DMA tile 0 into buffer 0 (wave 0 only; TDM ignores EXEC)
    if (tid < 32) {
        tdm_load_2d(asOff[0], a_tile, BK, BM, (unsigned long long)K);
        tdm_load_2d(bsOff[0], b_tile, BK, BN, (unsigned long long)K);
    }

    for (int t = 0; t < nsteps; ++t) {
        const int cur = t & 1;
        if (tid < 32) {
            if (t + 1 < nsteps) {
                const int nxt = cur ^ 1;
                const int k1 = (t + 1) * BK;
                tdm_load_2d(asOff[nxt], a_tile + k1, BK, BM, (unsigned long long)K);
                tdm_load_2d(bsOff[nxt], b_tile + k1, BK, BN, (unsigned long long)K);
                __builtin_amdgcn_s_wait_tensorcnt(2);  // tile t complete; t+1 in flight
            } else {
                __builtin_amdgcn_s_wait_tensorcnt(0);
            }
        }
        __syncthreads();

        // ---- fragment loads from LDS (ISA 16-bit A 16x32 / B 32x16 layouts) ----
        bf16x16 af[2], bfr[4];
#pragma unroll
        for (int i = 0; i < 2; ++i) {
            const bf16_t* p = &As[cur][(m_wave + i * 16 + fm) * LDSS + fh * 8];
            FragU fr;
            fr.u[0] = *(const uint4*)(p);        // K = h*8    .. h*8+7
            fr.u[1] = *(const uint4*)(p + 16);   // K = 16+h*8 .. 16+h*8+7
            af[i] = fr.f;
        }
#pragma unroll
        for (int j = 0; j < 4; ++j) {
            const bf16_t* p = &Bs[cur][(n_wave + j * 16 + fm) * LDSS + fh * 8];
            FragU fr;
            fr.u[0] = *(const uint4*)(p);        // K = h*8    .. h*8+7
            fr.u[1] = *(const uint4*)(p + 16);   // K = 16+h*8 .. 16+h*8+7
            bfr[j] = fr.f;
        }

        // ---- 8 WMMA per wave per K-step ----
#pragma unroll
        for (int i = 0; i < 2; ++i)
#pragma unroll
            for (int j = 0; j < 4; ++j)
                acc[i][j] = __builtin_amdgcn_wmma_f32_16x16x32_bf16(
                    false, af[i], false, bfr[j], (short)0, acc[i][j], false, false);

        __syncthreads();   // all waves done with buf[cur] before it is re-filled
    }

    // ---- store C (VGPR r: lane<16 -> M=r ; lane>=16 -> M=8+r) ----
#pragma unroll
    for (int i = 0; i < 2; ++i) {
#pragma unroll
        for (int j = 0; j < 4; ++j) {
            const int col  = n_wg + n_wave + j * 16 + fm;
            const int rowb = m_wg + m_wave + i * 16 + fh * 8;
#pragma unroll
            for (int r = 0; r < 8; ++r)
                C[(size_t)(rowb + r) * N + col] = (OutT)acc[i][j][r];
        }
    }
}

// ---------- fp32 -> bf16 cast ----------
__global__ __launch_bounds__(256)
void cast_f32_bf16(const float* __restrict__ in, bf16_t* __restrict__ out, long long n)
{
    const long long i = ((long long)blockIdx.x * 256 + threadIdx.x) * 4;
    if (i + 3 < n) {
        const float4 f = *(const float4*)(in + i);
        bf16x4 o;
        o.x = (bf16_t)f.x; o.y = (bf16_t)f.y; o.z = (bf16_t)f.z; o.w = (bf16_t)f.w;
        *(bf16x4*)(out + i) = o;
    }
}

// ---------- masked softmax row kernel: P = softmax(scale*S + mask), bf16 out ----------
__global__ __launch_bounds__(256)
void softmax_mask_kernel(const float* __restrict__ S, const int* __restrict__ mask,
                         bf16_t* __restrict__ P, float scale)
{
    const int n = blockIdx.x;       // query row
    const int b = blockIdx.y;       // batch
    const size_t rowoff = ((size_t)b * gridDim.x + n) * NK;
    const float* row  = S + rowoff;
    const int*   mrow = mask + (size_t)b * NK;
    bf16_t*      prow = P + rowoff;

    __shared__ float red[8];
    const int tid  = threadIdx.x;
    const int lane = tid & 31;
    const int wv   = tid >> 5;

    float v[4];
    float mx = -3.0e38f;
#pragma unroll
    for (int j = 0; j < 4; ++j) {
        const int m = tid + j * 256;
        const float s = row[m] * scale;
        v[j] = (mrow[m] != 0) ? s : -3.0e38f;
        mx = fmaxf(mx, v[j]);
    }
#pragma unroll
    for (int off = 16; off > 0; off >>= 1)
        mx = fmaxf(mx, __shfl_xor(mx, off, 32));
    if (lane == 0) red[wv] = mx;
    __syncthreads();
    float bmx = red[0];
#pragma unroll
    for (int i = 1; i < 8; ++i) bmx = fmaxf(bmx, red[i]);
    __syncthreads();

    float sum = 0.f;
#pragma unroll
    for (int j = 0; j < 4; ++j) {
        v[j] = __expf(v[j] - bmx);
        sum += v[j];
    }
#pragma unroll
    for (int off = 16; off > 0; off >>= 1)
        sum += __shfl_xor(sum, off, 32);
    if (lane == 0) red[wv] = sum;
    __syncthreads();
    float tot = 0.f;
#pragma unroll
    for (int i = 0; i < 8; ++i) tot += red[i];
    const float inv = 1.0f / tot;
#pragma unroll
    for (int j = 0; j < 4; ++j)
        prow[tid + j * 256] = (bf16_t)(v[j] * inv);
}

// ---------- bf16 transpose (per batch): out[Cc,R] = in[R,Cc]^T ----------
__global__ __launch_bounds__(256)
void transpose_bf16(const bf16_t* __restrict__ in, bf16_t* __restrict__ out, int R, int Cc)
{
    __shared__ bf16_t tile[32][33];
    const int b = blockIdx.z;
    const bf16_t* src = in  + (size_t)b * R * Cc;
    bf16_t*       dst = out + (size_t)b * R * Cc;
    const int c0 = blockIdx.x * 32;
    const int r0 = blockIdx.y * 32;
    const int tx = threadIdx.x & 31;
    const int ty = threadIdx.x >> 5;
#pragma unroll
    for (int i = 0; i < 32; i += 8)
        tile[ty + i][tx] = src[(size_t)(r0 + ty + i) * Cc + c0 + tx];
    __syncthreads();
#pragma unroll
    for (int i = 0; i < 32; i += 8)
        dst[(size_t)(c0 + ty + i) * R + r0 + tx] = tile[tx][ty + i];
}

// ---------- orchestration ----------
extern "C" void kernel_launch(void* const* d_in, const int* in_sizes, int n_in,
                              void* d_out, int out_size, void* d_ws, size_t ws_size,
                              hipStream_t stream)
{
    const float* q    = (const float*)d_in[0];   // [B, NQ, 1024]
    const float* k    = (const float*)d_in[1];   // [B, NK, 1024]
    const int*   mask = (const int*)  d_in[2];   // [B, NK]
    const float* wq   = (const float*)d_in[3];   // [E, 1024]
    const float* wk   = (const float*)d_in[4];   // [E, 1024]
    float* out = (float*)d_out;                  // [B, NQ, E]

    char* ws = (char*)d_ws;
    const size_t SZ_BF = (size_t)B_DIM * NQ * E_DIM * sizeof(bf16_t); // 64 MB
    const size_t SZ_W  = (size_t)E_DIM * E_DIM * sizeof(bf16_t);      //  2 MB

    bf16_t* qb  = (bf16_t*)(ws);                          // [B,NQ,D] bf16
    bf16_t* kb  = (bf16_t*)(ws + SZ_BF);                  // [B,NK,D] bf16
    bf16_t* wqb = (bf16_t*)(ws + 2 * SZ_BF);              // [E,D]    bf16
    bf16_t* wkb = (bf16_t*)(ws + 2 * SZ_BF + SZ_W);       // [E,D]    bf16
    bf16_t* qp  = (bf16_t*)(ws + 2 * SZ_BF + 2 * SZ_W);   // [B,NQ,E] bf16
    bf16_t* kp  = (bf16_t*)(ws + 3 * SZ_BF + 2 * SZ_W);   // [B,NK,E] bf16
    bf16_t* kpt = (bf16_t*)(ws + 4 * SZ_BF + 2 * SZ_W);   // [B,E,NK] bf16
    bf16_t* P   = (bf16_t*)(ws + 5 * SZ_BF + 2 * SZ_W);   // [B,NQ,NK] bf16
    float*  S   = (float*)(ws);   // [B,NQ,NK] f32, aliases qb+kb (dead by then)

    // 1) casts to bf16
    const long long nqk = (long long)B_DIM * NQ * E_DIM;
    const long long nw  = (long long)E_DIM * E_DIM;
    cast_f32_bf16<<<(unsigned)(nqk / 1024), 256, 0, stream>>>(q, qb, nqk);
    cast_f32_bf16<<<(unsigned)(nqk / 1024), 256, 0, stream>>>(k, kb, nqk);
    cast_f32_bf16<<<(unsigned)(nw / 1024), 256, 0, stream>>>(wq, wqb, nw);
    cast_f32_bf16<<<(unsigned)(nw / 1024), 256, 0, stream>>>(wk, wkb, nw);

    const long long sAB = (long long)NQ * E_DIM;      // per-batch matrix stride
    const dim3 gg(E_DIM / BN, NQ / BM, B_DIM);

    // 2) projections: qp = q @ wq^T ; kp = k @ wk^T   (weights: batch stride 0)
    gemm_nt_wmma<bf16_t><<<gg, 256, 0, stream>>>(qb, wqb, qp, NQ, E_DIM, E_DIM, sAB, 0, sAB);
    gemm_nt_wmma<bf16_t><<<gg, 256, 0, stream>>>(kb, wkb, kp, NK, E_DIM, E_DIM, sAB, 0, sAB);

    // 3) scores: S = qp @ kp^T (f32)
    gemm_nt_wmma<float><<<dim3(NK / BN, NQ / BM, B_DIM), 256, 0, stream>>>(
        qp, kp, S, NQ, NK, E_DIM, sAB, sAB, (long long)NQ * NK);

    // 4) masked softmax (scale = 1/sqrt(1024) = 0.03125), P bf16
    softmax_mask_kernel<<<dim3(NQ, B_DIM), 256, 0, stream>>>(S, mask, P, 0.03125f);

    // 5) kpt = kp^T per batch  ->  x = P @ kp becomes NT GEMM with kpt
    transpose_bf16<<<dim3(E_DIM / 32, NK / 32, B_DIM), 256, 0, stream>>>(kp, kpt, NK, E_DIM);

    // 6) out = P @ kpt^T (f32 out)
    gemm_nt_wmma<float><<<gg, 256, 0, stream>>>(
        P, kpt, out, NQ, E_DIM, NK, (long long)NQ * NK, sAB, sAB);
}
